// RegionSelector_86242943303789
// MI455X (gfx1250) — matrix operand order repaired and checked by reading, not compiled, not run on hardware
//
#include <hip/hip_runtime.h>
#include <cstdint>

// ---------------------------------------------------------------------------
// Problem constants (from reference)
// ---------------------------------------------------------------------------
#define BB   64
#define SS   196
#define DD   2048
#define HH   8
#define DKK  256
#define MM   16384
#define TK   32
#define CC   29

#define USE_TDM 1   // 1: tensor_load_to_lds staging; 0: per-lane async-load staging

typedef __bf16 bf16;
typedef __attribute__((ext_vector_type(8)))  __bf16 v8bf;
typedef __attribute__((ext_vector_type(16))) __bf16 v16bf;
typedef __attribute__((ext_vector_type(8)))  float  v8f;
typedef unsigned int u32;
typedef __attribute__((ext_vector_type(4))) u32 u32x4;
typedef __attribute__((ext_vector_type(4))) int i32x4;
typedef __attribute__((ext_vector_type(8))) int i32x8;

// ---------------------------------------------------------------------------
// fp32 -> bf16 convert
// ---------------------------------------------------------------------------
__global__ __launch_bounds__(256)
void conv_k(const float* __restrict__ s, bf16* __restrict__ d, long long n) {
    long long i = (long long)blockIdx.x * 256 + threadIdx.x;
    if (i < n) d[i] = (bf16)s[i];
}

// fp32 -> bf16 transpose-convert: dst[c][r] = src[r][c]; dst is cols x rows
__global__ __launch_bounds__(256)
void tconv_k(const float* __restrict__ src, bf16* __restrict__ dst,
             int rows, int cols) {
    __shared__ float t[32][33];
    int bx = blockIdx.x * 32, by = blockIdx.y * 32;
    int tx = threadIdx.x, ty = threadIdx.y; // block (32,8)
#pragma unroll
    for (int i = 0; i < 4; ++i) {
        int rr = by + ty + i * 8;
        t[ty + i * 8][tx] = src[(long long)rr * cols + bx + tx];
    }
    __syncthreads();
#pragma unroll
    for (int i = 0; i < 4; ++i) {
        int cc = bx + ty + i * 8;
        dst[(long long)cc * rows + by + tx] = (bf16)t[tx][ty + i * 8];
    }
}

// ---------------------------------------------------------------------------
// xm[b][d] = mean_s x[b][s][d]   (fp32 in, bf16 out)
// ---------------------------------------------------------------------------
__global__ __launch_bounds__(256)
void mean_k(const float* __restrict__ x, bf16* __restrict__ xm) {
    int d = blockIdx.x * 256 + threadIdx.x;
    int b = blockIdx.y;
    const float* px = x + ((long long)b * SS) * DD + d;
    float s = 0.f;
    for (int i = 0; i < SS; ++i) s += px[(long long)i * DD];
    xm[b * DD + d] = (bf16)(s * (1.0f / (float)SS));
}

// out[i] = sum_e A[i][e]*v[e] + b[i]     (bv' = wv @ memproj_b + bv)
__global__ __launch_bounds__(256)
void matvec_k(const float* __restrict__ A, const float* __restrict__ v,
              const float* __restrict__ b, float* __restrict__ out, int K) {
    __shared__ float red[256];
    int row = blockIdx.x;
    float s = 0.f;
    for (int e = threadIdx.x; e < K; e += 256)
        s += A[(long long)row * K + e] * v[e];
    red[threadIdx.x] = s; __syncthreads();
    for (int st = 128; st > 0; st >>= 1) {
        if ((int)threadIdx.x < st) red[threadIdx.x] += red[threadIdx.x + st];
        __syncthreads();
    }
    if (threadIdx.x == 0) out[row] = red[0] + b[row];
}

// ---------------------------------------------------------------------------
// LDS-tiled, double-buffered NT bf16 WMMA GEMM:
//   C[M,N] = scale*(A[M,K] . B[N,K]^T) + bias[N]   (optional exact-GELU)
// Block: 256 threads (8 waves), tile 32x256, K-step 64.
// Global->LDS staging: Tensor Data Mover (TENSOR_LOAD_TO_LDS, one descriptor
// per tile, TENSORcnt double-buffered) or per-lane GLOBAL_LOAD_ASYNC_TO_LDS.
// LDS rows use a 144-byte pitch (TDM pad: 32 DWORDs interval / 4 DWORDs pad)
// -> 36-bank stride, conflict-free ds_load fragments.
// Fragment layouts per CDNA5 ISA 7.12.2 (16-bit A 16x32, B 32x16, f32 C/D).
// Requires: M%32==0, N%256==0, K%64==0.
// ---------------------------------------------------------------------------
#define GPITCH 72   // bf16 elements per LDS tile row (64 + 8 pad)

#if USE_TDM
// One TDM descriptor: load tile (tile1 rows x 64 cols, bf16) from a row-major
// matrix with row pitch `stride0` elements into LDS at `lds_addr`, padding
// each 128B row with 16B (-> GPITCH). D# packing per ISA 08_async_tensor §8.
__device__ __forceinline__ void tdm_load_2d(const bf16* g, u32 lds_addr,
                                            int tile1, long long stride0) {
    unsigned long long ga = (unsigned long long)(uintptr_t)g;
    u32x4 g0;
    g0[0] = 1u;                                   // count=1, user mode
    g0[1] = lds_addr;                             // LDS byte address
    g0[2] = (u32)ga;                              // global addr [31:0]
    g0[3] = ((u32)(ga >> 32) & 0x01FFFFFFu) | (2u << 30);  // [56:32] | type=2
    i32x8 g1;
    // data_size=2B, pad_enable, pad_interval: 32 DWORDs (code 4),
    // pad_amount: 4 DWORDs (code 3); workgroup_mask=0 (not in a cluster)
    g1[0] = (int)((1u << 16) | (1u << 20) | (4u << 22) | (3u << 25));
    g1[1] = (int)(64u << 16);                     // tensor_dim0 = 64 (= tile)
    g1[2] = (int)((u32)tile1 << 16);              // tensor_dim1 = tile1
    g1[3] = (int)(64u << 16);                     // tile_dim0 = 64
    g1[4] = tile1 & 0xFFFF;                       // tile_dim1 (tile_dim2 = 0)
    g1[5] = (int)(u32)stride0;                    // tensor_dim0_stride lo
    g1[6] = (int)((u32)(stride0 >> 32) & 0xFFFFu);// stride hi | dim1_stride=0
    g1[7] = 0;
    i32x4 gz = {0, 0, 0, 0};                      // 2D: groups 2/3 unused
#if defined(__clang_major__) && __clang_major__ >= 23
    i32x8 gz8 = {0, 0, 0, 0, 0, 0, 0, 0};
    __builtin_amdgcn_tensor_load_to_lds(g0, g1, gz, gz, gz8, 0);
#else
    __builtin_amdgcn_tensor_load_to_lds(g0, g1, gz, gz, 0);
#endif
}
#else
__device__ __forceinline__ void async_b128(const bf16* g, const bf16* l) {
    unsigned lds = (unsigned)(uintptr_t)l;
    unsigned long long ga = (unsigned long long)(uintptr_t)g;
    asm volatile("global_load_async_to_lds_b128 %0, %1, off"
                 :: "v"(lds), "v"(ga) : "memory");
}
#endif

template <typename OutT, bool GELU>
__global__ __launch_bounds__(256)
void gemm_nt(const bf16* __restrict__ A, int lda, long long sA,
             const bf16* __restrict__ Bm, int ldb, long long sB,
             OutT* __restrict__ C, int ldc, long long sC,
             const float* __restrict__ bias, long long sBias,
             float scale, int K) {
    extern __shared__ bf16 smem[];
    bf16* aBuf0 = smem;                       // 32*GPITCH
    bf16* aBuf1 = smem + 32 * GPITCH;
    bf16* bBuf0 = smem + 64 * GPITCH;         // 256*GPITCH
    bf16* bBuf1 = smem + 64 * GPITCH + 256 * GPITCH;

    int tid  = threadIdx.x;
    int lane = tid & 31;
    int wave = tid >> 5;
    int z = blockIdx.z;
    A  += (long long)z * sA;
    Bm += (long long)z * sB;
    C  += (long long)z * sC;
    if (bias) bias += (long long)z * sBias;

    int mblock = blockIdx.y * 32;
    int nblock = blockIdx.x * 256;
    int r  = lane & 15;     // A row / B column within 16-tile
    int hi = lane >> 4;     // K-half selector
    int mi = wave >> 2;     // which of the 2 M sub-tiles
    int nq = wave & 3;      // which 64-wide N strip

#if !USE_TDM
    int arow = tid >> 3, aoff = (tid & 7) * 8;   // per-thread 16B chunk coords
    const bf16* gA = A + (long long)(mblock + arow) * lda + aoff;
#endif

    auto issue_tile = [&](int k0, int buf) {
        bf16* aB = buf ? aBuf1 : aBuf0;
        bf16* bB = buf ? bBuf1 : bBuf0;
#if USE_TDM
        if (wave == 0) {   // one wave drives the DMA engine (EXEC ignored)
            tdm_load_2d(A  + (long long)mblock * lda + k0,
                        (u32)(uintptr_t)aB, 32,  (long long)lda);
            tdm_load_2d(Bm + (long long)nblock * ldb + k0,
                        (u32)(uintptr_t)bB, 256, (long long)ldb);
        }
#else
        async_b128(gA + k0, aB + arow * GPITCH + aoff);
#pragma unroll
        for (int i = 0; i < 8; ++i) {
            int c = tid + i * 256;
            int row = c >> 3, off = (c & 7) * 8;
            async_b128(Bm + (long long)(nblock + row) * ldb + k0 + off,
                       bB + row * GPITCH + off);
        }
#endif
    };

    v8f acc[4] = {};
    int nt = K / 64;
    issue_tile(0, 0);
    for (int kt = 0; kt < nt; ++kt) {
        // stage next tile, then wait until the CURRENT tile is resident:
        // per-wave tensor/async ops complete in order, so <= (ops of next
        // tile) outstanding implies the previous tile has landed in LDS.
#if USE_TDM
        if (kt + 1 < nt) {
            issue_tile((kt + 1) * 64, (kt + 1) & 1);
            if (wave == 0) __builtin_amdgcn_s_wait_tensorcnt(2);
        } else {
            if (wave == 0) __builtin_amdgcn_s_wait_tensorcnt(0);
        }
#else
        if (kt + 1 < nt) {
            issue_tile((kt + 1) * 64, (kt + 1) & 1);
            asm volatile("s_wait_asynccnt %0" :: "n"(9) : "memory");
        } else {
            asm volatile("s_wait_asynccnt %0" :: "n"(0) : "memory");
        }
#endif
        __syncthreads();

        const bf16* aP = (kt & 1 ? aBuf1 : aBuf0) + (mi * 16 + r) * GPITCH + hi * 8;
        const bf16* bP = (kt & 1 ? bBuf1 : bBuf0) + (nq * 64 + r) * GPITCH + hi * 8;

        // batch all fragment loads, then burst the 8 WMMAs (staged dscnt)
        v16bf afr[2];
        v16bf bfr[2][4];
#pragma unroll
        for (int k2 = 0; k2 < 2; ++k2) {
            int kl = k2 * 32;
            v8bf a0 = *(const v8bf*)(aP + kl);
            v8bf a1 = *(const v8bf*)(aP + kl + 16);
            afr[k2] = __builtin_shufflevector(a0, a1,
                          0,1,2,3,4,5,6,7,8,9,10,11,12,13,14,15);
#pragma unroll
            for (int t = 0; t < 4; ++t) {
                v8bf b0 = *(const v8bf*)(bP + t * 16 * GPITCH + kl);
                v8bf b1 = *(const v8bf*)(bP + t * 16 * GPITCH + kl + 16);
                bfr[k2][t] = __builtin_shufflevector(b0, b1,
                                 0,1,2,3,4,5,6,7,8,9,10,11,12,13,14,15);
            }
        }
#pragma unroll
        for (int k2 = 0; k2 < 2; ++k2)
#pragma unroll
            for (int t = 0; t < 4; ++t)
                acc[t] = __builtin_amdgcn_wmma_f32_16x16x32_bf16(
                             false, afr[k2], false, bfr[k2][t], (short)0,
                             acc[t], false, false);
        __syncthreads();   // tile fully consumed before its buffer is refilled
    }

    int mtile = mblock + mi * 16 + hi * 8;
    int nbase = nblock + nq * 64;
#pragma unroll
    for (int t = 0; t < 4; ++t) {
        int ncol = nbase + t * 16 + r;
        float bv = bias ? bias[ncol] : 0.f;
#pragma unroll
        for (int j = 0; j < 8; ++j) {
            float v = acc[t][j] * scale + bv;
            if (GELU) v = 0.5f * v * (1.f + erff(v * 0.70710678118f));
            C[(long long)(mtile + j) * ldc + ncol] = (OutT)v;
        }
    }
}
#define GEMM_LDS ((64 * GPITCH + 512 * GPITCH) * 2)   // 82944 bytes

// ---------------------------------------------------------------------------
// Top-32 + softmax per score row (16384 wide). Row staged in 64KB LDS;
// 32 extract-max passes with block reduction; wave32 shuffle softmax.
// Row id r = h*B + b (h-major), matching qproj/wmem/attn layout.
// ---------------------------------------------------------------------------
__global__ __launch_bounds__(256)
void topk_softmax_k(const float* __restrict__ scores,
                    float* __restrict__ p, int* __restrict__ idxo) {
    extern __shared__ float srow[];          // MM floats
    float* rv = srow + MM;                   // 256
    int*   ri = (int*)(rv + 256);            // 256
    __shared__ float sel[TK];
    int rrow = blockIdx.x, tid = threadIdx.x;
    const float* S = scores + (long long)rrow * MM;
    for (int i = tid; i < MM; i += 256) srow[i] = S[i];
    __syncthreads();
    for (int k = 0; k < TK; ++k) {
        float bv = -3.4e38f; int bi = 0x7fffffff;
        for (int i = tid; i < MM; i += 256) {
            float v = srow[i];
            if (v > bv || (v == bv && i < bi)) { bv = v; bi = i; }
        }
        rv[tid] = bv; ri[tid] = bi; __syncthreads();
        for (int s = 128; s > 0; s >>= 1) {
            if (tid < s) {
                float ov = rv[tid + s]; int oi = ri[tid + s];
                if (ov > rv[tid] || (ov == rv[tid] && oi < ri[tid])) {
                    rv[tid] = ov; ri[tid] = oi;
                }
            }
            __syncthreads();
        }
        if (tid == 0) {
            sel[k] = rv[0];
            idxo[rrow * TK + k] = ri[0];
            srow[ri[0]] = -3.4e38f;           // exclude from later passes
        }
        __syncthreads();
    }
    if (tid < TK) {                           // one wave32 does the softmax
        float e = __expf(sel[tid] - sel[0]);  // sel[0] is the max
        float s = e;
#pragma unroll
        for (int m = 16; m > 0; m >>= 1) s += __shfl_xor(s, m, 32);
        p[rrow * TK + tid] = e / s;
    }
}

// wmem[r][e] = sum_k p[r][k] * mem_bf[idx[r][k]][e]   (softmax-weighted mem)
__global__ __launch_bounds__(256)
void wmem_k(const bf16* __restrict__ mem_bf, const float* __restrict__ p,
            const int* __restrict__ idxo, bf16* __restrict__ wmem) {
    int r = blockIdx.x;
    int e = blockIdx.y * 256 + threadIdx.x;
    const float* pp = p + r * TK;
    const int*   ii = idxo + r * TK;
    float acc = 0.f;
#pragma unroll 4
    for (int k = 0; k < TK; ++k)
        acc += pp[k] * (float)mem_bf[(long long)ii[k] * DD + e];
    wmem[(long long)r * DD + e] = (bf16)acc;
}

// logits[b][j] = sum_e fused_bf[b][e]*head_w[j][e] + head_b[j]   (64x29, tiny)
__global__ __launch_bounds__(128)
void logits_k(const bf16* __restrict__ fused, const float* __restrict__ hw,
              const float* __restrict__ hb, float* __restrict__ out) {
    __shared__ float red[128];
    int j = blockIdx.x, b = blockIdx.y;
    float s = 0.f;
    for (int e = threadIdx.x; e < DD; e += 128)
        s += (float)fused[b * DD + e] * hw[(long long)j * DD + e];
    red[threadIdx.x] = s; __syncthreads();
    for (int st = 64; st > 0; st >>= 1) {
        if ((int)threadIdx.x < st) red[threadIdx.x] += red[threadIdx.x + st];
        __syncthreads();
    }
    if (threadIdx.x == 0) out[b * CC + j] = red[0] + hb[j];
}

// ---------------------------------------------------------------------------
// Host orchestration
// ---------------------------------------------------------------------------
extern "C" void kernel_launch(void* const* d_in, const int* in_sizes, int n_in,
                              void* d_out, int out_size, void* d_ws, size_t ws_size,
                              hipStream_t stream) {
    const float* x         = (const float*)d_in[0];
    // d_in[1] box_masks: unused by the reference math
    const float* mem       = (const float*)d_in[2];
    const float* ff_w      = (const float*)d_in[3];
    const float* ff_b      = (const float*)d_in[4];
    const float* memproj_w = (const float*)d_in[5];
    const float* memproj_b = (const float*)d_in[6];
    const float* wq        = (const float*)d_in[7];
    const float* bq        = (const float*)d_in[8];
    const float* wk        = (const float*)d_in[9];
    // d_in[10] bk: contributes a per-(b,h) constant to scores -> top-k/softmax invariant
    const float* wv        = (const float*)d_in[11];
    const float* bv        = (const float*)d_in[12];
    const float* wo        = (const float*)d_in[13];
    const float* bo        = (const float*)d_in[14];
    const float* fuse_w    = (const float*)d_in[15];
    const float* fuse_b    = (const float*)d_in[16];
    const float* head_w    = (const float*)d_in[17];
    const float* head_b    = (const float*)d_in[18];
    float* out = (float*)d_out;

    // ---- workspace bump allocator ----
    char* w = (char*)d_ws;
    size_t off = 0;
    auto alloc = [&](size_t bytes) -> void* {
        void* pp = (void*)(w + off);
        off += (bytes + 255) & ~(size_t)255;
        return pp;
    };
    bf16* mem_bf    = (bf16*)alloc((size_t)MM * DD * 2);   // 67 MB
    bf16* mpT       = (bf16*)alloc((size_t)DD * DD * 2);   // memproj_w^T
    bf16* wkT       = (bf16*)alloc((size_t)DD * DD * 2);   // wk^T
    bf16* wv_bf     = (bf16*)alloc((size_t)DD * DD * 2);
    bf16* ffw_bf    = (bf16*)alloc((size_t)DD * DD * 2);
    bf16* wq_bf     = (bf16*)alloc((size_t)DD * DD * 2);
    bf16* wo_bf     = (bf16*)alloc((size_t)DD * DD * 2);
    bf16* fusew_bf  = (bf16*)alloc((size_t)DD * 2 * DD * 2);
    bf16* wvp_bf    = (bf16*)alloc((size_t)DD * DD * 2);   // Wv' = wv@memproj_w
    bf16* xm_bf     = (bf16*)alloc((size_t)BB * DD * 2);
    bf16* q_bf      = (bf16*)alloc((size_t)BB * DD * 2);
    bf16* t_bf      = (bf16*)alloc((size_t)HH * BB * DD * 2);
    bf16* qproj_bf  = (bf16*)alloc((size_t)HH * BB * DD * 2);
    float* scores   = (float*)alloc((size_t)HH * BB * MM * 4); // 33.5 MB
    float* pbuf     = (float*)alloc((size_t)HH * BB * TK * 4);
    int*   idxb     = (int*)alloc((size_t)HH * BB * TK * 4);
    bf16* wmem_bf   = (bf16*)alloc((size_t)HH * BB * DD * 2);
    bf16* attn_bf   = (bf16*)alloc((size_t)BB * DD * 2);
    bf16* fusedin_bf= (bf16*)alloc((size_t)BB * 2 * DD * 2);
    bf16* fused_bf  = (bf16*)alloc((size_t)BB * DD * 2);
    float* bvp      = (float*)alloc((size_t)DD * 4);       // wv@memproj_b + bv
    (void)ws_size; (void)in_sizes; (void)n_in; (void)out_size;

    dim3 blk256(256), blk128(128), blkT(32, 8);

    // ---- Phase A: conversions / reductions ----
    conv_k<<<dim3((MM * DD) / 256), blk256, 0, stream>>>(mem, mem_bf, (long long)MM * DD);
    tconv_k<<<dim3(DD / 32, DD / 32), blkT, 0, stream>>>(memproj_w, mpT, DD, DD);
    tconv_k<<<dim3(DD / 32, DD / 32), blkT, 0, stream>>>(wk, wkT, DD, DD);
    conv_k<<<dim3((DD * DD) / 256), blk256, 0, stream>>>(wv, wv_bf, (long long)DD * DD);
    conv_k<<<dim3((DD * DD) / 256), blk256, 0, stream>>>(ff_w, ffw_bf, (long long)DD * DD);
    conv_k<<<dim3((DD * DD) / 256), blk256, 0, stream>>>(wq, wq_bf, (long long)DD * DD);
    conv_k<<<dim3((DD * DD) / 256), blk256, 0, stream>>>(wo, wo_bf, (long long)DD * DD);
    conv_k<<<dim3((DD * 2 * DD) / 256), blk256, 0, stream>>>(fuse_w, fusew_bf, (long long)DD * 2 * DD);
    mean_k<<<dim3(DD / 256, BB), blk256, 0, stream>>>(x, xm_bf);
    matvec_k<<<dim3(DD), blk256, 0, stream>>>(wv, memproj_b, bv, bvp, DD);

    // ---- Phase B: WMMA GEMM chain (LDS-tiled, TDM-staged) ----
    // xf = xm @ ff_w^T + ff_b  -> left half of fusedin (ldc = 4096)
    gemm_nt<bf16, false><<<dim3(DD / 256, BB / 32, 1), blk256, GEMM_LDS, stream>>>(
        xm_bf, DD, 0, ffw_bf, DD, 0, fusedin_bf, 2 * DD, 0, ff_b, 0, 1.f, DD);
    // q = xf @ wq^T + bq
    gemm_nt<bf16, false><<<dim3(DD / 256, BB / 32, 1), blk256, GEMM_LDS, stream>>>(
        fusedin_bf, 2 * DD, 0, wq_bf, DD, 0, q_bf, DD, 0, bq, 0, 1.f, DD);
    // t[h,b,:] = q_h (64x256) @ wk_h (256x2048)   [fold q through wk, per head]
    gemm_nt<bf16, false><<<dim3(DD / 256, BB / 32, HH), blk256, GEMM_LDS, stream>>>(
        q_bf, DD, DKK, wkT, DD, DKK, t_bf, DD, (long long)BB * DD,
        nullptr, 0, 1.f, DKK);
    // qproj = t (512x2048) @ memproj_w (2048x2048)   [fold through memproj]
    gemm_nt<bf16, false><<<dim3(DD / 256, (HH * BB) / 32, 1), blk256, GEMM_LDS, stream>>>(
        t_bf, DD, 0, mpT, DD, 0, qproj_bf, DD, 0, nullptr, 0, 1.f, DD);
    // Wv' = wv @ memproj_w (2048^3)
    gemm_nt<bf16, false><<<dim3(DD / 256, DD / 32, 1), blk256, GEMM_LDS, stream>>>(
        wv_bf, DD, 0, mpT, DD, 0, wvp_bf, DD, 0, nullptr, 0, 1.f, DD);
    // scores = (qproj @ mem^T) / sqrt(DK)    (512 x 16384, the big one)
    gemm_nt<float, false><<<dim3(MM / 256, (HH * BB) / 32, 1), blk256, GEMM_LDS, stream>>>(
        qproj_bf, DD, 0, mem_bf, DD, 0, scores, MM, 0, nullptr, 0, 0.0625f, DD);
    // top-32 + softmax per row
    topk_softmax_k<<<dim3(HH * BB), blk256, MM * 4 + 256 * 8, stream>>>(
        scores, pbuf, idxb);
    // softmax-weighted memory rows
    wmem_k<<<dim3(HH * BB, DD / 256), blk256, 0, stream>>>(mem_bf, pbuf, idxb, wmem_bf);
    // attn[b, h*256+d] = wmem[h,b,:] @ Wv'_h^T + bv'_h   (per-head 64x256x2048)
    gemm_nt<bf16, false><<<dim3(DKK / 256, BB / 32, HH), blk256, GEMM_LDS, stream>>>(
        wmem_bf, DD, (long long)BB * DD, wvp_bf, DD, (long long)DKK * DD,
        attn_bf, DD, DKK, bvp, DKK, 1.f, DD);
    // resp = attn @ wo^T + bo  -> right half of fusedin
    gemm_nt<bf16, false><<<dim3(DD / 256, BB / 32, 1), blk256, GEMM_LDS, stream>>>(
        attn_bf, DD, 0, wo_bf, DD, 0, fusedin_bf + DD, 2 * DD, 0, bo, 0, 1.f, DD);
    // fused = gelu([xf, resp] @ fuse_w^T + fuse_b)   (K = 4096)
    gemm_nt<bf16, true><<<dim3(DD / 256, BB / 32, 1), blk256, GEMM_LDS, stream>>>(
        fusedin_bf, 2 * DD, 0, fusew_bf, 2 * DD, 0, fused_bf, DD, 0,
        fuse_b, 0, 1.f, 2 * DD);
    // logits = fused @ head_w^T + head_b   (64x29, VALU)
    logits_k<<<dim3(CC, BB), blk128, 0, stream>>>(fused_bf, head_w, head_b, out);
}
